// Model_22960895165111
// MI455X (gfx1250) — compile-verified, matrix operand order checked by reading, not run on hardware
//
#include <hip/hip_runtime.h>
#include <hip/hip_bf16.h>

typedef _Float16 half_t;
typedef __attribute__((ext_vector_type(16))) _Float16 v16h;
typedef __attribute__((ext_vector_type(8)))  _Float16 v8h;
typedef __attribute__((ext_vector_type(8)))  float    v8f;

#define NTOK 2560
#define DIMV 256
#define NHEAD 8
#define HDIM 32
#define MLPD 1024
#define SCALE 0.17677669529663687f  /* 1/sqrt(32) */

// ---------------------------------------------------------------- fragment IO
// 16-bit A-fragment (16x32): lane l -> row (l&15); elements 0..7 = K[hf*8..+7],
// elements 8..15 = K[16+hf*8..+7], hf = l>>4.  B-fragments use the same
// pattern on a pre-transposed [n][k] buffer.
__device__ __forceinline__ v16h load_frag(const half_t* base, int stride_h, int lane){
  int r = lane & 15, hf = lane >> 4;
  const half_t* p = base + (size_t)r * stride_h + hf * 8;
  v8h lo = *(const v8h*)(p);
  v8h hi = *(const v8h*)(p + 16);
  v16h o;
#pragma unroll
  for (int i = 0; i < 8; ++i){ o[i] = lo[i]; o[i + 8] = hi[i]; }
  return o;
}

__device__ __forceinline__ v8f wmma16(v16h a, v16h b, v8f c){
  return __builtin_amdgcn_wmma_f32_16x16x32_f16(false, a, false, b, (short)0, c, false, false);
}

__device__ __forceinline__ v8f zero8(){
  v8f z;
#pragma unroll
  for (int i = 0; i < 8; ++i) z[i] = 0.0f;
  return z;
}

// ---------------------------------------------------------------- layernorm
__global__ __launch_bounds__(256) void layernorm_f16(
    const float* __restrict__ x, const float* __restrict__ g,
    const float* __restrict__ b, half_t* __restrict__ out)
{
  __shared__ float s1[256];
  __shared__ float s2[256];
  int row = blockIdx.x, tid = threadIdx.x;
  float v = x[(size_t)row * DIMV + tid];
  s1[tid] = v; s2[tid] = v * v;
  __syncthreads();
  for (int s = 128; s > 0; s >>= 1){
    if (tid < s){ s1[tid] += s1[tid + s]; s2[tid] += s2[tid + s]; }
    __syncthreads();
  }
  float mu  = s1[0] * (1.0f / DIMV);
  float var = s2[0] * (1.0f / DIMV) - mu * mu;
  float r   = rsqrtf(var + 1e-5f);
  out[(size_t)row * DIMV + tid] = (half_t)((v - mu) * r * g[tid] + b[tid]);
}

// ------------------------------------------------- weight f32 -> f16, transposed
// W[k][n] (f32, K x Nn)  ->  Wt[n][k] (f16)
__global__ void wtrans(const float* __restrict__ W, half_t* __restrict__ Wt,
                       int K, int Nn)
{
  int idx = blockIdx.x * 256 + threadIdx.x;
  if (idx >= K * Nn) return;
  int k = idx % K, n = idx / K;
  Wt[idx] = (half_t)W[(size_t)k * Nn + n];
}

// --------------------------------------------------------------- v transpose
// qkv[j][512 + h*32 + t] -> vt[h][t][j]
__global__ void transpose_v(const half_t* __restrict__ qkv, half_t* __restrict__ vt)
{
  int idx = blockIdx.x * 256 + threadIdx.x;       // 8*32*2560 total, exact grid
  int j = idx % NTOK;
  int t = (idx / NTOK) & 31;
  int h = idx / (NTOK * 32);
  vt[idx] = qkv[(size_t)j * 768 + 2 * DIMV + h * HDIM + t];
}

// ------------------------------------------------------------------- GEMM
// C[M][Nn] = A[M][K](f16) * Bt[Nn][K](f16)^T   (128x64 block, 8 waves, WMMA)
// mode 0: out_h = C                       (f16 store)
// mode 1: accum += C + bias               (f32 residual accumulate)
// mode 2: out_h = gelu_exact(C + bias)    (f16 store)
__global__ __launch_bounds__(256) void gemm_wmma(
    const half_t* __restrict__ A, const half_t* __restrict__ Bt,
    int M, int K, int Nn, const float* __restrict__ bias,
    half_t* __restrict__ out_h, float* __restrict__ accum, int mode)
{
  __shared__ __align__(16) half_t As[128 * 40];
  __shared__ __align__(16) half_t Bs[64 * 40];
  int tid = threadIdx.x, lane = tid & 31, w = tid >> 5;
  int m0 = blockIdx.x * 128, n0 = blockIdx.y * 64;

  v8f acc[4];
#pragma unroll
  for (int t = 0; t < 4; ++t) acc[t] = zero8();

  for (int k0 = 0; k0 < K; k0 += 32){
    __syncthreads();
#pragma unroll
    for (int idx = tid; idx < 512; idx += 256){        // A: 128 rows x 4 chunks
      int r = idx >> 2, c = (idx & 3) * 8;
      *(v8h*)(&As[r * 40 + c]) = *(const v8h*)(&A[(size_t)(m0 + r) * K + k0 + c]);
    }
    {                                                  // B: 64 rows x 4 chunks
      int r = tid >> 2, c = (tid & 3) * 8;
      *(v8h*)(&Bs[r * 40 + c]) = *(const v8h*)(&Bt[(size_t)(n0 + r) * K + k0 + c]);
    }
    __syncthreads();
    v16h af = load_frag(&As[w * 16 * 40], 40, lane);
#pragma unroll
    for (int t = 0; t < 4; ++t){
      v16h bf = load_frag(&Bs[t * 16 * 40], 40, lane);
      acc[t] = wmma16(af, bf, acc[t]);
    }
  }

  int hf = lane >> 4, nn = lane & 15;
#pragma unroll
  for (int t = 0; t < 4; ++t){
    int n = n0 + t * 16 + nn;
    float bv = bias ? bias[n] : 0.0f;
#pragma unroll
    for (int v = 0; v < 8; ++v){
      int m = m0 + w * 16 + 8 * hf + v;
      float val = acc[t][v];
      if (mode == 0){
        out_h[(size_t)m * Nn + n] = (half_t)val;
      } else if (mode == 1){
        accum[(size_t)m * Nn + n] += val + bv;
      } else {
        val += bv;
        val = 0.5f * val * (1.0f + erff(val * 0.70710678118654752f));
        out_h[(size_t)m * Nn + n] = (half_t)val;
      }
    }
  }
}

// ------------------------------------------------- attention pass 1: scores
// One wave per (head, 16-row q tile).  Writes raw scaled scores (0 in masked
// blocks) to attn and per-row online-softmax stats (max, sum) to ws.
__global__ __launch_bounds__(256) void attn_scores(
    const half_t* __restrict__ qkv, const int* __restrict__ mask,
    float* __restrict__ attn, float* __restrict__ row_m, float* __restrict__ row_l)
{
  int tid = threadIdx.x, lane = tid & 31, w = tid >> 5;
  int head = blockIdx.x;
  int i0 = blockIdx.y * 128 + w * 16;
  int hf = lane >> 4, nn = lane & 15;
  int ib = i0 >> 9;

  bool allow[5];
#pragma unroll
  for (int jb = 0; jb < 4; ++jb) allow[jb] = (jb == ib) || (ib == 4 && mask[jb] != 0);
  allow[4] = (ib == 4);

  v16h qf = load_frag(qkv + (size_t)i0 * 768 + head * HDIM, 768, lane);

  float mx[8], ls[8];
#pragma unroll
  for (int v = 0; v < 8; ++v){ mx[v] = -3.0e38f; ls[v] = 0.0f; }

  size_t base = (size_t)head * NTOK * NTOK;
  for (int j0 = 0; j0 < NTOK; j0 += 16){
    size_t rowbase = base + (size_t)(i0 + 8 * hf) * NTOK + j0 + nn;
    if (!allow[j0 >> 9]){
#pragma unroll
      for (int v = 0; v < 8; ++v) attn[rowbase + (size_t)v * NTOK] = 0.0f;
      continue;
    }
    v16h kf = load_frag(qkv + (size_t)j0 * 768 + DIMV + head * HDIM, 768, lane);
    v8f s = wmma16(qf, kf, zero8());
#pragma unroll
    for (int v = 0; v < 8; ++v){
      float sv = s[v] * SCALE;
      attn[rowbase + (size_t)v * NTOK] = sv;
      if (sv > mx[v]){ ls[v] = ls[v] * __expf(mx[v] - sv) + 1.0f; mx[v] = sv; }
      else           { ls[v] += __expf(sv - mx[v]); }
    }
  }
  // combine stats across the 16 lanes that share each row
#pragma unroll
  for (int off = 1; off < 16; off <<= 1){
#pragma unroll
    for (int v = 0; v < 8; ++v){
      float om = __shfl_xor(mx[v], off, 32);
      float ol = __shfl_xor(ls[v], off, 32);
      float nm = fmaxf(mx[v], om);
      ls[v] = ls[v] * __expf(mx[v] - nm) + ol * __expf(om - nm);
      mx[v] = nm;
    }
  }
  if (nn == 0){
#pragma unroll
    for (int v = 0; v < 8; ++v){
      int i = i0 + 8 * hf + v;
      row_m[head * NTOK + i] = mx[v];
      row_l[head * NTOK + i] = ls[v];
    }
  }
}

// ------------------------------------- attention pass 2: normalize + attn@V
__global__ __launch_bounds__(256) void attn_norm_av(
    const half_t* __restrict__ vt, const int* __restrict__ mask,
    float* __restrict__ attn, const float* __restrict__ row_m,
    const float* __restrict__ row_l, half_t* __restrict__ att_out)
{
  __shared__ __align__(16) half_t Ap[8][16 * 40];   // per-wave 16x32 p tile
  int tid = threadIdx.x, lane = tid & 31, w = tid >> 5;
  int head = blockIdx.x;
  int i0 = blockIdx.y * 128 + w * 16;
  int hf = lane >> 4, nn = lane & 15;
  int ib = i0 >> 9;

  bool allow[5];
#pragma unroll
  for (int jb = 0; jb < 4; ++jb) allow[jb] = (jb == ib) || (ib == 4 && mask[jb] != 0);
  allow[4] = (ib == 4);

  float rm[8], ri[8];
#pragma unroll
  for (int v = 0; v < 8; ++v){
    int i = i0 + 8 * hf + v;
    rm[v] = row_m[head * NTOK + i];
    ri[v] = 1.0f / row_l[head * NTOK + i];
  }

  v8f o0 = zero8(), o1 = zero8();
  half_t* ap = &Ap[w][0];
  const half_t* vth = vt + (size_t)head * HDIM * NTOK;
  size_t base = (size_t)head * NTOK * NTOK;

  for (int j0 = 0; j0 < NTOK; j0 += 32){
    if (!allow[j0 >> 9]) continue;
#pragma unroll
    for (int sub = 0; sub < 2; ++sub){
      size_t rb = base + (size_t)(i0 + 8 * hf) * NTOK + j0 + sub * 16 + nn;
#pragma unroll
      for (int v = 0; v < 8; ++v){
        float s = attn[rb + (size_t)v * NTOK];
        float p = __expf(s - rm[v]) * ri[v];
        attn[rb + (size_t)v * NTOK] = p;
        ap[(8 * hf + v) * 40 + sub * 16 + nn] = (half_t)p;
      }
    }
    asm volatile("s_wait_dscnt 0x0" ::: "memory");     // wave-local LDS RAW
    v16h af = load_frag(ap, 40, lane);
    v16h b0 = load_frag(vth + j0, NTOK, lane);                       // t = 0..15
    v16h b1 = load_frag(vth + (size_t)16 * NTOK + j0, NTOK, lane);   // t = 16..31
    o0 = wmma16(af, b0, o0);
    o1 = wmma16(af, b1, o1);
  }
#pragma unroll
  for (int v = 0; v < 8; ++v){
    int i = i0 + 8 * hf + v;
    att_out[(size_t)i * DIMV + head * HDIM + nn]      = (half_t)o0[v];
    att_out[(size_t)i * DIMV + head * HDIM + 16 + nn] = (half_t)o1[v];
  }
}

// =============================================================== launch
extern "C" void kernel_launch(void* const* d_in, const int* in_sizes, int n_in,
                              void* d_out, int out_size, void* d_ws, size_t ws_size,
                              hipStream_t stream)
{
  const float* x_in  = (const float*)d_in[0];
  const int*   maskp = (const int*)  d_in[1];
  const float* ln1_g = (const float*)d_in[2];
  const float* ln1_b = (const float*)d_in[3];
  const float* Wqkv  = (const float*)d_in[4];
  const float* Wproj = (const float*)d_in[5];
  const float* bproj = (const float*)d_in[6];
  const float* ln2_g = (const float*)d_in[7];
  const float* ln2_b = (const float*)d_in[8];
  const float* W1    = (const float*)d_in[9];
  const float* b1    = (const float*)d_in[10];
  const float* W2    = (const float*)d_in[11];
  const float* b2    = (const float*)d_in[12];

  char* wsp = (char*)d_ws;
  auto carve = [&](size_t bytes) -> char* {
    char* p = wsp; wsp += (bytes + 255) & ~(size_t)255; return p;
  };
  float*  x_ws   = (float*) carve((size_t)NTOK * DIMV * 4);
  half_t* h16    = (half_t*)carve((size_t)NTOK * DIMV * 2);
  half_t* qkv16  = (half_t*)carve((size_t)NTOK * 768 * 2);
  half_t* vt16   = (half_t*)carve((size_t)NHEAD * HDIM * NTOK * 2);
  half_t* ao16   = (half_t*)carve((size_t)NTOK * DIMV * 2);
  half_t* mlp16  = (half_t*)carve((size_t)NTOK * MLPD * 2);
  float*  row_m  = (float*) carve((size_t)NHEAD * NTOK * 4);
  float*  row_l  = (float*) carve((size_t)NHEAD * NTOK * 4);
  half_t* Wqkv_t = (half_t*)carve((size_t)768 * DIMV * 2);
  half_t* Wproj_t= (half_t*)carve((size_t)DIMV * DIMV * 2);
  half_t* W1_t   = (half_t*)carve((size_t)MLPD * DIMV * 2);
  half_t* W2_t   = (half_t*)carve((size_t)DIMV * MLPD * 2);

  hipMemcpyAsync(x_ws, x_in, (size_t)NTOK * DIMV * 4, hipMemcpyDeviceToDevice, stream);

  for (int d = 0; d < 2; ++d){
    float* attn_d = (float*)d_out + (size_t)NTOK * DIMV
                  + (size_t)d * NHEAD * NTOK * NTOK;

    wtrans<<<(DIMV * 768  + 255) / 256, 256, 0, stream>>>(Wqkv  + (size_t)d * DIMV * 768,  Wqkv_t,  DIMV, 768);
    wtrans<<<(DIMV * DIMV + 255) / 256, 256, 0, stream>>>(Wproj + (size_t)d * DIMV * DIMV, Wproj_t, DIMV, DIMV);
    wtrans<<<(DIMV * MLPD + 255) / 256, 256, 0, stream>>>(W1    + (size_t)d * DIMV * MLPD, W1_t,    DIMV, MLPD);
    wtrans<<<(MLPD * DIMV + 255) / 256, 256, 0, stream>>>(W2    + (size_t)d * MLPD * DIMV, W2_t,    MLPD, DIMV);

    layernorm_f16<<<NTOK, 256, 0, stream>>>(x_ws, ln1_g + d * DIMV, ln1_b + d * DIMV, h16);

    gemm_wmma<<<dim3(NTOK / 128, 768 / 64), 256, 0, stream>>>(
        h16, Wqkv_t, NTOK, DIMV, 768, nullptr, qkv16, nullptr, 0);

    transpose_v<<<(NHEAD * HDIM * NTOK) / 256, 256, 0, stream>>>(qkv16, vt16);

    attn_scores<<<dim3(NHEAD, NTOK / 128), 256, 0, stream>>>(
        qkv16, maskp, attn_d, row_m, row_l);

    attn_norm_av<<<dim3(NHEAD, NTOK / 128), 256, 0, stream>>>(
        vt16, maskp, attn_d, row_m, row_l, ao16);

    gemm_wmma<<<dim3(NTOK / 128, DIMV / 64), 256, 0, stream>>>(
        ao16, Wproj_t, NTOK, DIMV, DIMV, bproj + d * DIMV, nullptr, x_ws, 1);

    layernorm_f16<<<NTOK, 256, 0, stream>>>(x_ws, ln2_g + d * DIMV, ln2_b + d * DIMV, h16);

    gemm_wmma<<<dim3(NTOK / 128, MLPD / 64), 256, 0, stream>>>(
        h16, W1_t, NTOK, DIMV, MLPD, b1 + d * MLPD, mlp16, nullptr, 2);

    gemm_wmma<<<dim3(NTOK / 128, DIMV / 64), 256, 0, stream>>>(
        mlp16, W2_t, NTOK, MLPD, DIMV, b2 + d * DIMV, nullptr, x_ws, 1);
  }

  hipMemcpyAsync(d_out, x_ws, (size_t)NTOK * DIMV * 4, hipMemcpyDeviceToDevice, stream);
}
